// ContinousActionDecoder_77025943487062
// MI455X (gfx1250) — matrix operand order repaired and checked by reading, not compiled, not run on hardware
//
#include <hip/hip_runtime.h>
#include <hip/hip_bf16.h>

// ---------------------------------------------------------------------------
// Nearest-neighbor action decode:
//   argmin_n |x_b - a_n|^2  ==  argmin_n ( a2[n] - 2 * x_b . a_n )
// GEMM runs on v_wmma_f32_16x16x32_bf16 with an fp32->bf16 hi/lo split
// (3 bf16 WMMAs per tile ~ fp32 accuracy).
// Wave tile: 32(M) x 64(N) -> 24 WMMAs per 24 b128 loads per K-step.
// ---------------------------------------------------------------------------

static constexpr int Bsz = 4096;
static constexpr int Nsz = 65536;
static constexpr int Dsz = 512;

typedef __attribute__((ext_vector_type(16))) __bf16 v16bf;
typedef __attribute__((ext_vector_type(8)))  float  v8f;

union Frag16 { uint4 q[2]; v16bf v; };

__device__ __forceinline__ v8f wmma_bf16(v16bf a, v16bf b, v8f c) {
    return __builtin_amdgcn_wmma_f32_16x16x32_bf16(
        /*neg_a=*/false, a, /*neg_b=*/false, b,
        /*c_mod=*/(short)0, c, /*reuse_a=*/false, /*reuse_b=*/false);
}

// Monotone float -> uint mapping so integer min == float min.
__device__ __forceinline__ unsigned fkey(float f) {
    unsigned u = __float_as_uint(f);
    return (u & 0x80000000u) ? ~u : (u | 0x80000000u);
}

// ---------------------------------------------------------------------------
// Prep: split f32 rows into bf16 hi/lo; optionally compute row norms |a_n|^2.
// One wave per row (512 elems = 16 per lane).
// ---------------------------------------------------------------------------
__global__ __launch_bounds__(256)
void prep_split(const float* __restrict__ src, __bf16* __restrict__ hi,
                __bf16* __restrict__ lo, float* __restrict__ norms, int rows) {
    const int lane = threadIdx.x & 31;
    const int wave = threadIdx.x >> 5;
    const int row  = blockIdx.x * 8 + wave;
    if (row >= rows) return;

    const float* s = src + (size_t)row * Dsz;
    __bf16* hrow = hi + (size_t)row * Dsz;
    __bf16* lrow = lo + (size_t)row * Dsz;

    float sum = 0.0f;
#pragma unroll
    for (int j = 0; j < 4; ++j) {
        const int e = lane * 16 + j * 4;
        float4 v = *(const float4*)(s + e);
        union { __bf16 h[4]; uint2 u; } H, L;
        H.h[0] = (__bf16)v.x; L.h[0] = (__bf16)(v.x - (float)H.h[0]);
        H.h[1] = (__bf16)v.y; L.h[1] = (__bf16)(v.y - (float)H.h[1]);
        H.h[2] = (__bf16)v.z; L.h[2] = (__bf16)(v.z - (float)H.h[2]);
        H.h[3] = (__bf16)v.w; L.h[3] = (__bf16)(v.w - (float)H.h[3]);
        *(uint2*)(hrow + e) = H.u;
        *(uint2*)(lrow + e) = L.u;
        sum = fmaf(v.x, v.x, sum);
        sum = fmaf(v.y, v.y, sum);
        sum = fmaf(v.z, v.z, sum);
        sum = fmaf(v.w, v.w, sum);
    }
    if (norms != nullptr) {
#pragma unroll
        for (int m = 1; m < 32; m <<= 1) sum += __shfl_xor(sum, m);
        if (lane == 0) norms[row] = sum;
    }
}

__global__ void init_best(unsigned long long* best, int n) {
    int i = blockIdx.x * blockDim.x + threadIdx.x;
    if (i < n) best[i] = ~0ull;
}

// ---------------------------------------------------------------------------
// Fused GEMM + row argmin.
// Block: 256 threads = 8 waves, covers 32 (M) x 512 (N).
// Wave: 32 (M) x 64 (N): 2 M-tiles x 4 N-tiles, 3 bf16 WMMAs per tile pair
// per K-step (24 WMMAs, 24 b128 loads). A fragments reused 4x, B fragments 6x.
// Grid: x = M tiles (innermost -> consecutive blocks share the 1 MB action
// chunk in L2 while sweeping the 8 MB pred matrix).
// ---------------------------------------------------------------------------
__global__ __launch_bounds__(256)
void gemm_argmin(const __bf16* __restrict__ xh, const __bf16* __restrict__ xl,
                 const __bf16* __restrict__ ah, const __bf16* __restrict__ al,
                 const float* __restrict__ a2,
                 unsigned long long* __restrict__ best) {
    const int lane  = threadIdx.x & 31;
    const int wave  = threadIdx.x >> 5;      // 8 N strips per block
    const int mBase = blockIdx.x * 32;       // wave owns both 16-row tiles
    const int nBase = blockIdx.y * 512 + wave * 64;
    const int half  = lane >> 4;             // 0: lanes 0-15, 1: lanes 16-31
    const int r     = lane & 15;

    const __bf16* xh0 = xh + (size_t)(mBase + r) * Dsz;        // M-tile 0
    const __bf16* xl0 = xl + (size_t)(mBase + r) * Dsz;
    const __bf16* xh1 = xh + (size_t)(mBase + 16 + r) * Dsz;   // M-tile 1
    const __bf16* xl1 = xl + (size_t)(mBase + 16 + r) * Dsz;

    v8f acc[2][4];
#pragma unroll
    for (int mt = 0; mt < 2; ++mt)
#pragma unroll
        for (int t = 0; t < 4; ++t)
#pragma unroll
            for (int i = 0; i < 8; ++i) acc[mt][t][i] = 0.0f;

    const int koffA = half * 8;    // A fragment: chunks at K=koffA, K=koffA+16
    const int koffB = half * 8;    // B fragment: chunks at K=koffB, K=koffB+16

    for (int k0 = 0; k0 < Dsz; k0 += 32) {
        Frag16 fah0, fal0, fah1, fal1;
        fah0.q[0] = *(const uint4*)(xh0 + k0 + koffA);
        fah0.q[1] = *(const uint4*)(xh0 + k0 + koffA + 16);
        fal0.q[0] = *(const uint4*)(xl0 + k0 + koffA);
        fal0.q[1] = *(const uint4*)(xl0 + k0 + koffA + 16);
        fah1.q[0] = *(const uint4*)(xh1 + k0 + koffA);
        fah1.q[1] = *(const uint4*)(xh1 + k0 + koffA + 16);
        fal1.q[0] = *(const uint4*)(xl1 + k0 + koffA);
        fal1.q[1] = *(const uint4*)(xl1 + k0 + koffA + 16);
#pragma unroll
        for (int t = 0; t < 4; ++t) {
            const size_t rowoff = (size_t)(nBase + t * 16 + r) * Dsz + k0 + koffB;
            const __bf16* bh = ah + rowoff;
            const __bf16* bl = al + rowoff;
            Frag16 fbh, fbl;
            fbh.q[0] = *(const uint4*)(bh);
            fbh.q[1] = *(const uint4*)(bh + 16);
            fbl.q[0] = *(const uint4*)(bl);
            fbl.q[1] = *(const uint4*)(bl + 16);
            // prefetch next K chunk of this action row toward the WGP
            __builtin_prefetch(bh + 32, 0, 3);
            acc[0][t] = wmma_bf16(fah0.v, fbh.v, acc[0][t]);   // hi*hi (M0)
            acc[0][t] = wmma_bf16(fah0.v, fbl.v, acc[0][t]);   // hi*lo
            acc[0][t] = wmma_bf16(fal0.v, fbh.v, acc[0][t]);   // lo*hi
            acc[1][t] = wmma_bf16(fah1.v, fbh.v, acc[1][t]);   // hi*hi (M1)
            acc[1][t] = wmma_bf16(fah1.v, fbl.v, acc[1][t]);   // hi*lo
            acc[1][t] = wmma_bf16(fal1.v, fbh.v, acc[1][t]);   // lo*hi
        }
    }

    // Epilogue: score = a2[n] - 2 * dot; per-row argmin.
    // C layout: lane holds column n = nBase + t*16 + r; element i is row
    // M = mBase + mt*16 + half*8 + i.
#pragma unroll
    for (int mt = 0; mt < 2; ++mt) {
        float bv[8];
        int   bn[8];
#pragma unroll
        for (int i = 0; i < 8; ++i) { bv[i] = 3.4e38f; bn[i] = 0; }

#pragma unroll
        for (int t = 0; t < 4; ++t) {
            const int n = nBase + t * 16 + r;
            const float a2n = a2[n];
#pragma unroll
            for (int i = 0; i < 8; ++i) {
                const float v = fmaf(-2.0f, acc[mt][t][i], a2n);
                if (v < bv[i] || (v == bv[i] && n < bn[i])) { bv[i] = v; bn[i] = n; }
            }
        }
        // reduce across the 16 lanes of this half-wave
#pragma unroll
        for (int m = 1; m < 16; m <<= 1) {
#pragma unroll
            for (int i = 0; i < 8; ++i) {
                const float ov = __shfl_xor(bv[i], m);
                const int   on = __shfl_xor(bn[i], m);
                if (ov < bv[i] || (ov == bv[i] && on < bn[i])) { bv[i] = ov; bn[i] = on; }
            }
        }
        if (r == 0) {
#pragma unroll
            for (int i = 0; i < 8; ++i) {
                const int row = mBase + mt * 16 + half * 8 + i;
                const unsigned long long key =
                    ((unsigned long long)fkey(bv[i]) << 32) | (unsigned)bn[i];
                atomicMin(best + row, key);
            }
        }
    }
}

// ---------------------------------------------------------------------------
// Gather the winning action rows (f32) into the output.
// ---------------------------------------------------------------------------
__global__ __launch_bounds__(128)
void gather_rows(const float* __restrict__ act,
                 const unsigned long long* __restrict__ best,
                 float* __restrict__ out) {
    const int row = blockIdx.x;
    const int idx = (int)(best[row] & 0xffffffffull);
    const float4* s = (const float4*)(act + (size_t)idx * Dsz);
    float4* d = (float4*)(out + (size_t)row * Dsz);
    for (int e = threadIdx.x; e < Dsz / 4; e += blockDim.x) d[e] = s[e];
}

// ---------------------------------------------------------------------------
extern "C" void kernel_launch(void* const* d_in, const int* in_sizes, int n_in,
                              void* d_out, int out_size, void* d_ws, size_t ws_size,
                              hipStream_t stream) {
    const float* pred = (const float*)d_in[0];   // [B, D] f32
    const float* act  = (const float*)d_in[1];   // [N, D] f32
    float* out = (float*)d_out;                  // [B, D] f32

    // Workspace layout (all power-of-two sizes, naturally aligned):
    char* ws = (char*)d_ws;
    size_t off = 0;
    __bf16* a_hi = (__bf16*)(ws + off); off += (size_t)Nsz * Dsz * sizeof(__bf16);
    __bf16* a_lo = (__bf16*)(ws + off); off += (size_t)Nsz * Dsz * sizeof(__bf16);
    __bf16* x_hi = (__bf16*)(ws + off); off += (size_t)Bsz * Dsz * sizeof(__bf16);
    __bf16* x_lo = (__bf16*)(ws + off); off += (size_t)Bsz * Dsz * sizeof(__bf16);
    float*  a2   = (float*)(ws + off);  off += (size_t)Nsz * sizeof(float);
    unsigned long long* best = (unsigned long long*)(ws + off);
    off += (size_t)Bsz * sizeof(unsigned long long);

    prep_split<<<Nsz / 8, 256, 0, stream>>>(act,  a_hi, a_lo, a2,      Nsz);
    prep_split<<<Bsz / 8, 256, 0, stream>>>(pred, x_hi, x_lo, nullptr, Bsz);
    init_best<<<(Bsz + 255) / 256, 256, 0, stream>>>(best, Bsz);

    dim3 grid(Bsz / 32, Nsz / 512);   // x = M tiles innermost for L2 reuse
    gemm_argmin<<<grid, 256, 0, stream>>>(x_hi, x_lo, a_hi, a_lo, a2, best);

    gather_rows<<<Bsz, 128, 0, stream>>>(act, best, out);
}
